// StrucFeaGNN_46076409151515
// MI455X (gfx1250) — matrix operand order repaired and checked by reading, not compiled, not run on hardware
//
#include <hip/hip_runtime.h>
#include <hip/hip_bf16.h>
#include <math.h>

typedef __attribute__((ext_vector_type(16))) _Float16 v16h;
typedef __attribute__((ext_vector_type(2)))  __fp16   v2fp16;
typedef __attribute__((ext_vector_type(8)))  float    v8f;

#define NNODES 100000
#define NEDGES 3200000
#define NGRAPH 128
#define DIN    1024
#define HDIM   64

union Frag {
    v16h h;
    unsigned u[8];
    uint4 q[2];
};

// single-instruction packed f32x2 -> f16x2 (v_cvt_pk_rtz_f16_f32)
__device__ __forceinline__ unsigned pack2h(float a, float b) {
    union { v2fp16 h; unsigned u; } p;
    p.h = __builtin_amdgcn_cvt_pkrtz(a, b);
    return p.u;
}

// ---------------------------------------------------------------- zero fill
__global__ void sfg_zero_f32(float* __restrict__ p, int n) {
    int i = blockIdx.x * blockDim.x + threadIdx.x;
    int stride = gridDim.x * blockDim.x;
    for (; i < n; i += stride) p[i] = 0.0f;
}

// --------------------------------------------------- prep: pre3_w B-fragments
// B tile (K=32 x N=16) per-lane layout used here (contiguous k, same order as the A tile):
//   lanes 0-15  : column N=lane,    K=0..15  packed in pairs across 8 dwords
//   lanes 16-31 : column N=lane-16, K=16..31 packed in pairs across 8 dwords
// w3 is (16 x 1022) row-major; Bmat[k][n] = w3[n][k], zero-padded to K=1024.
__global__ void sfg_prep_w3(const float* __restrict__ w3, unsigned* __restrict__ w3f) {
    int kt   = blockIdx.x;          // 32 K-tiles of 32
    int lane = threadIdx.x;         // 32 lanes
    int n    = lane & 15;
    int koff = (lane < 16) ? 0 : 16;
#pragma unroll
    for (int d = 0; d < 8; ++d) {
        int k0 = kt * 32 + koff + 2 * d;
        float a = (k0     < 1022) ? w3[n * 1022 + k0]     : 0.0f;
        float b = (k0 + 1 < 1022) ? w3[n * 1022 + k0 + 1] : 0.0f;
        w3f[(kt * 32 + lane) * 8 + d] = pack2h(a, b);
    }
}

// -------------------------------------------- prep: 64x64 GIN weight fragments
// frag layout: [nt(4)][kt(2)][lane(32)][8 dwords]; Bmat[k][n]=w[nt*16+n][kt*32+k]
__global__ void sfg_prep_gin(const float* __restrict__ w, unsigned* __restrict__ frag) {
    int b    = blockIdx.x;          // 8 blocks = nt*2+kt
    int nt   = b >> 1;
    int kt   = b & 1;
    int lane = threadIdx.x;
    int n    = nt * 16 + (lane & 15);
    int koff = kt * 32 + ((lane < 16) ? 0 : 16);
#pragma unroll
    for (int d = 0; d < 8; ++d) {
        int k0 = koff + 2 * d;
        frag[((nt * 2 + kt) * 32 + lane) * 8 + d] =
            pack2h(w[n * 64 + k0], w[n * 64 + k0 + 1]);
    }
}

// ------------------------------------------- prep: fused BN scale/bias vectors
// layer block of 256 floats: [0..63]=s1, [64..127]=bb1, [128..191]=s2, [192..255]=bb2
__global__ void sfg_prep_fused(const float* b1_0, const float* bng0, const float* bnb0,
                               const float* b2_0, const float* obng0, const float* obnb0,
                               const float* b1_1, const float* bng1, const float* bnb1,
                               const float* b2_1, const float* obng1, const float* obnb1,
                               float* __restrict__ fused) {
    int t = threadIdx.x;            // 128 threads
    if (t >= 128) return;
    int layer = t >> 6;
    int j = t & 63;
    const float S = rsqrtf(1.0f + 1e-5f);
    const float* b1  = layer ? b1_1  : b1_0;
    const float* bg  = layer ? bng1  : bng0;
    const float* bb  = layer ? bnb1  : bnb0;
    const float* b2  = layer ? b2_1  : b2_0;
    const float* og  = layer ? obng1 : obng0;
    const float* ob  = layer ? obnb1 : obnb0;
    float s1  = bg[j] * S;
    float bb1 = b1[j] * s1 + bb[j];
    float s2  = og[j] * S;
    float bb2 = b2[j] * s2 + ob[j];
    float* f = fused + layer * 256;
    f[j]       = s1;
    f[64 + j]  = bb1;
    f[128 + j] = s2;
    f[192 + j] = bb2;
}

// --------------------------------------------------------------- pre-MLP stage
// One wave handles 16 nodes. WMMA over K=1024 (padded) for ident@pre3_w.T with a
// double-buffered LDS A-tile (one barrier per K-tile), then the tiny
// 2->16->32 and 16->32 MLPs per lane using LDS-staged weights.
#define P4W 0
#define P2W 512
#define P4B 1024
#define P2B 1056
#define P1W 1088
#define P1B 1120
#define WLN 1136

__global__ __launch_bounds__(32)
void sfg_pre_mlp(const float* __restrict__ x,
                 const float* __restrict__ p1w, const float* __restrict__ p1b,
                 const float* __restrict__ p2w, const float* __restrict__ p2b,
                 const float* __restrict__ p3b,
                 const float* __restrict__ p4w, const float* __restrict__ p4b,
                 const unsigned* __restrict__ w3f,
                 float* __restrict__ newx) {
    __shared__ unsigned AtU[2][16 * 16]; // double-buffered 16 nodes x 32 halfs
    __shared__ float    ix16s[16][17];
    __shared__ float    wl[WLN];         // staged stage-2 weights

    const int tile = blockIdx.x;
    const int lane = threadIdx.x;
    const int r  = lane & 15;
    const int h2 = lane >> 4;

    // stage stage-2 weights into LDS (visibility covered by in-loop barriers)
    for (int i = lane; i < 512; i += 32) {
        wl[P4W + i] = p4w[i];
        wl[P2W + i] = p2w[i];
    }
    wl[P4B + lane] = p4b[lane];
    wl[P2B + lane] = p2b[lane];
    wl[P1W + lane] = p1w[lane];
    if (lane < 16) wl[P1B + lane] = p1b[lane];

    const float4* xrow = (const float4*)(x + (size_t)(tile * 16 + r) * DIN);

    v8f c = {};
    for (int kt = 0; kt < 32; ++kt) {
        unsigned* buf = AtU[kt & 1];
#pragma unroll
        for (int q = 0; q < 4; ++q) {
            float4 f = xrow[kt * 8 + h2 * 4 + q];
            buf[r * 16 + h2 * 8 + q * 2 + 0] = pack2h(f.x, f.y);
            buf[r * 16 + h2 * 8 + q * 2 + 1] = pack2h(f.z, f.w);
        }
        if (kt < 31) __builtin_prefetch(xrow + (kt + 1) * 8 + h2 * 4, 0, 0);
        __syncthreads();

        Frag a, b;
        {
            const unsigned* rowp = &buf[(lane & 15) * 16];
            int o = (lane >> 4) * 8;
#pragma unroll
            for (int d = 0; d < 4; ++d) {
                a.u[d]     = rowp[o + d];
                a.u[4 + d] = rowp[o + 4 + d];
            }
        }
        const uint4* bq = (const uint4*)(w3f + (size_t)(kt * 32 + lane) * 8);
        b.q[0] = bq[0];
        b.q[1] = bq[1];
        c = __builtin_amdgcn_wmma_f32_16x16x32_f16(false, a.h, false, b.h,
                                                   (short)0, c, false, false);
    }

    // bias + relu on accumulator, spill to LDS (C layout: col=lane&15, row=i+8*(lane>>4))
    {
        int feat = lane & 15;
        int mb   = (lane >> 4) * 8;
        float bias = p3b[feat];
#pragma unroll
        for (int i = 0; i < 8; ++i)
            ix16s[mb + i][feat] = fmaxf(c[i] + bias, 0.0f);
    }
    __syncthreads();

    // second stage: lane -> (node = lane&15, output half = lane>>4)
    {
        int node = lane & 15;
        int oh   = lane >> 4;
        size_t nb = (size_t)(tile * 16 + node);

        float a0 = x[nb * DIN + 1022];
        float a1 = x[nb * DIN + 1023];
        float s16v[16];
#pragma unroll
        for (int j = 0; j < 16; ++j)
            s16v[j] = fmaxf(wl[P1W + j * 2] * a0 + wl[P1W + j * 2 + 1] * a1
                            + wl[P1B + j], 0.0f);

        float* orow = newx + nb * HDIM;
#pragma unroll
        for (int o = 0; o < 16; ++o) {
            int og = oh * 16 + o;
            float si = wl[P2B + og], ii = wl[P4B + og];
#pragma unroll
            for (int k = 0; k < 16; ++k) {
                si += s16v[k]        * wl[P2W + og * 16 + k];
                ii += ix16s[node][k] * wl[P4W + og * 16 + k];
            }
            orow[og]      = fmaxf(ii, 0.0f);   // cols 0..31 = ix path
            orow[32 + og] = fmaxf(si, 0.0f);   // cols 32..63 = struct path
        }
    }
}

// ---------------------------------------------------- edge scatter segment-sum
__global__ void sfg_scatter(const float* __restrict__ h,
                            const int* __restrict__ ei,
                            float* __restrict__ agg) {
    int gid = blockIdx.x * blockDim.x + threadIdx.x;
    if (gid >= NEDGES * 16) return;
    int e = gid >> 4;
    int l = gid & 15;
    int src = ei[e];
    int dst = ei[NEDGES + e];
    float4 v = ((const float4*)h)[(size_t)src * 16 + l];
    float* d = agg + (size_t)dst * HDIM + l * 4;
    atomicAdd(d + 0, v.x);
    atomicAdd(d + 1, v.y);
    atomicAdd(d + 2, v.z);
    atomicAdd(d + 3, v.w);
}

// ------------------------------------------------------- GIN dense (per layer)
// z = hin + agg; y = relu(s1*(z@W1^T)+bb1); out = s2*(y@W2^T)+bb2 + res1 (+res2)
__global__ __launch_bounds__(32)
void sfg_gin_dense(const float* __restrict__ hin, const float* __restrict__ agg,
                   const float* __restrict__ res1, const float* __restrict__ res2,
                   const unsigned* __restrict__ w1f, const unsigned* __restrict__ w2f,
                   const float* __restrict__ fl,    // fused 256-float block
                   float* __restrict__ hout) {
    __shared__ unsigned AtU[16 * 32];   // 16 x 64 halfs
    __shared__ unsigned YtU[16 * 32];

    const int tile = blockIdx.x;
    const int lane = threadIdx.x;
    const int r  = lane & 15;
    const int h2 = lane >> 4;

    // load z = hin + agg, convert to f16 tile
    {
        const float4* hr = (const float4*)(hin + (size_t)(tile * 16 + r) * HDIM);
        const float4* gr = (const float4*)(agg + (size_t)(tile * 16 + r) * HDIM);
#pragma unroll
        for (int q = 0; q < 8; ++q) {
            float4 a = hr[h2 * 8 + q];
            float4 g = gr[h2 * 8 + q];
            AtU[r * 32 + h2 * 16 + q * 2 + 0] = pack2h(a.x + g.x, a.y + g.y);
            AtU[r * 32 + h2 * 16 + q * 2 + 1] = pack2h(a.z + g.z, a.w + g.w);
        }
    }
    __syncthreads();

    const int feat0 = lane & 15;
    const int mb    = (lane >> 4) * 8;
    const int o     = (lane >> 4) * 8;

    Frag a[2];
#pragma unroll
    for (int kt = 0; kt < 2; ++kt) {
        const unsigned* rowp = &AtU[(lane & 15) * 32 + kt * 16];
#pragma unroll
        for (int d = 0; d < 4; ++d) {
            a[kt].u[d]     = rowp[o + d];
            a[kt].u[4 + d] = rowp[o + 4 + d];
        }
    }

    _Float16* Yt = (_Float16*)YtU;
#pragma unroll
    for (int nt = 0; nt < 4; ++nt) {
        v8f c = {};
#pragma unroll
        for (int kt = 0; kt < 2; ++kt) {
            Frag b;
            const uint4* bq = (const uint4*)(w1f + (size_t)((nt * 2 + kt) * 32 + lane) * 8);
            b.q[0] = bq[0];
            b.q[1] = bq[1];
            c = __builtin_amdgcn_wmma_f32_16x16x32_f16(false, a[kt].h, false, b.h,
                                                       (short)0, c, false, false);
        }
        int feat = nt * 16 + feat0;
        float s1 = fl[feat], bb1 = fl[64 + feat];
#pragma unroll
        for (int i = 0; i < 8; ++i) {
            float y = fmaxf(c[i] * s1 + bb1, 0.0f);
            Yt[(mb + i) * 64 + feat] = (_Float16)y;
        }
    }
    __syncthreads();

    Frag a2[2];
#pragma unroll
    for (int kt = 0; kt < 2; ++kt) {
        const unsigned* rowp = &YtU[(lane & 15) * 32 + kt * 16];
#pragma unroll
        for (int d = 0; d < 4; ++d) {
            a2[kt].u[d]     = rowp[o + d];
            a2[kt].u[4 + d] = rowp[o + 4 + d];
        }
    }

#pragma unroll
    for (int nt = 0; nt < 4; ++nt) {
        v8f c = {};
#pragma unroll
        for (int kt = 0; kt < 2; ++kt) {
            Frag b;
            const uint4* bq = (const uint4*)(w2f + (size_t)((nt * 2 + kt) * 32 + lane) * 8);
            b.q[0] = bq[0];
            b.q[1] = bq[1];
            c = __builtin_amdgcn_wmma_f32_16x16x32_f16(false, a2[kt].h, false, b.h,
                                                       (short)0, c, false, false);
        }
        int feat = nt * 16 + feat0;
        float s2 = fl[128 + feat], bb2 = fl[192 + feat];
#pragma unroll
        for (int i = 0; i < 8; ++i) {
            size_t addr = (size_t)(tile * 16 + mb + i) * HDIM + feat;
            float rsum = res1[addr];
            if (res2) rsum += res2[addr];
            hout[addr] = c[i] * s2 + bb2 + rsum;
        }
    }
}

// -------------------------------------------------------------- graph pooling
__global__ void sfg_pool(const float* __restrict__ h, const int* __restrict__ batch,
                         float* __restrict__ sums, float* __restrict__ cnts) {
    int gid = blockIdx.x * blockDim.x + threadIdx.x;
    if (gid >= NNODES * 16) return;
    int n = gid >> 4;
    int l = gid & 15;
    int b = batch[n];
    float4 v = ((const float4*)h)[(size_t)n * 16 + l];
    float* d = sums + (size_t)b * HDIM + l * 4;
    atomicAdd(d + 0, v.x);
    atomicAdd(d + 1, v.y);
    atomicAdd(d + 2, v.z);
    atomicAdd(d + 3, v.w);
    if (l == 0) atomicAdd(&cnts[b], 1.0f);
}

// -------------------------------------------------------- head MLP + logsoftmax
__global__ void sfg_head(const float* __restrict__ sums, const float* __restrict__ cnts,
                         const float* __restrict__ q1w, const float* __restrict__ q1b,
                         const float* __restrict__ q2w, const float* __restrict__ q2b,
                         float* __restrict__ out) {
    int g = blockIdx.x * blockDim.x + threadIdx.x;
    if (g >= NGRAPH) return;
    float inv = 1.0f / fmaxf(cnts[g], 1.0f);
    float pooled[64];
#pragma unroll
    for (int k = 0; k < 64; ++k) pooled[k] = sums[g * 64 + k] * inv;
    float h1[16];
#pragma unroll
    for (int j = 0; j < 16; ++j) {
        float acc = q1b[j];
#pragma unroll
        for (int k = 0; k < 64; ++k) acc += pooled[k] * q1w[j * 64 + k];
        h1[j] = fmaxf(acc, 0.0f);
    }
    float lg[7];
    float m = -1e30f;
#pragma unroll
    for (int oi = 0; oi < 7; ++oi) {
        float acc = q2b[oi];
#pragma unroll
        for (int j = 0; j < 16; ++j) acc += h1[j] * q2w[oi * 16 + j];
        lg[oi] = acc;
        m = fmaxf(m, acc);
    }
    float s = 0.0f;
#pragma unroll
    for (int oi = 0; oi < 7; ++oi) s += expf(lg[oi] - m);
    float lse = logf(s);
#pragma unroll
    for (int oi = 0; oi < 7; ++oi) out[g * 7 + oi] = lg[oi] - m - lse;
}

extern "C" void kernel_launch(void* const* d_in, const int* in_sizes, int n_in,
                              void* d_out, int out_size, void* d_ws, size_t ws_size,
                              hipStream_t stream) {
    (void)in_sizes; (void)n_in; (void)out_size; (void)ws_size;

    const float* x     = (const float*)d_in[0];
    const int*   ei    = (const int*)  d_in[1];
    const int*   batch = (const int*)  d_in[2];
    const float* p1w = (const float*)d_in[3];
    const float* p1b = (const float*)d_in[4];
    const float* p2w = (const float*)d_in[5];
    const float* p2b = (const float*)d_in[6];
    const float* p3w = (const float*)d_in[7];
    const float* p3b = (const float*)d_in[8];
    const float* p4w = (const float*)d_in[9];
    const float* p4b = (const float*)d_in[10];
    const float* q1w = (const float*)d_in[11];
    const float* q1b = (const float*)d_in[12];
    const float* q2w = (const float*)d_in[13];
    const float* q2b = (const float*)d_in[14];
    const float* g0w1  = (const float*)d_in[15];
    const float* g0b1  = (const float*)d_in[16];
    const float* g0bng = (const float*)d_in[17];
    const float* g0bnb = (const float*)d_in[18];
    const float* g0w2  = (const float*)d_in[19];
    const float* g0b2  = (const float*)d_in[20];
    const float* bn0g  = (const float*)d_in[21];
    const float* bn0b  = (const float*)d_in[22];
    const float* g1w1  = (const float*)d_in[23];
    const float* g1b1  = (const float*)d_in[24];
    const float* g1bng = (const float*)d_in[25];
    const float* g1bnb = (const float*)d_in[26];
    const float* g1w2  = (const float*)d_in[27];
    const float* g1b2  = (const float*)d_in[28];
    const float* bn1g  = (const float*)d_in[29];
    const float* bn1b  = (const float*)d_in[30];
    float* out = (float*)d_out;

    // workspace layout (floats)
    const size_t HB = (size_t)NNODES * HDIM;          // 6.4M floats per buffer
    float* newx = (float*)d_ws;
    float* g0   = newx + HB;
    float* g1   = g0 + HB;
    float* agg  = g1 + HB;
    unsigned* w3f  = (unsigned*)(agg + HB);           // 32*32*8 = 8192
    unsigned* w1f0 = w3f + 8192;                      // 4*2*32*8 = 2048 each
    unsigned* w2f0 = w1f0 + 2048;
    unsigned* w1f1 = w2f0 + 2048;
    unsigned* w2f1 = w1f1 + 2048;
    float* fused = (float*)(w2f1 + 2048);             // 512 floats
    float* sums  = fused + 512;                       // 128*64
    float* cnts  = sums + NGRAPH * HDIM;              // 128

    const int TILES = NNODES / 16;                    // 6250

    // weight prep
    sfg_prep_w3 <<<32, 32, 0, stream>>>(p3w, w3f);
    sfg_prep_gin<<<8, 32, 0, stream>>>(g0w1, w1f0);
    sfg_prep_gin<<<8, 32, 0, stream>>>(g0w2, w2f0);
    sfg_prep_gin<<<8, 32, 0, stream>>>(g1w1, w1f1);
    sfg_prep_gin<<<8, 32, 0, stream>>>(g1w2, w2f1);
    sfg_prep_fused<<<1, 128, 0, stream>>>(g0b1, g0bng, g0bnb, g0b2, bn0g, bn0b,
                                          g1b1, g1bng, g1bnb, g1b2, bn1g, bn1b, fused);

    // pre-MLP -> new_x
    sfg_pre_mlp<<<TILES, 32, 0, stream>>>(x, p1w, p1b, p2w, p2b, p3b, p4w, p4b,
                                          w3f, newx);

    // GIN layer 0
    sfg_zero_f32<<<4096, 256, 0, stream>>>(agg, (int)HB);
    sfg_scatter <<<(NEDGES * 16) / 256, 256, 0, stream>>>(newx, ei, agg);
    sfg_gin_dense<<<TILES, 32, 0, stream>>>(newx, agg, newx, nullptr,
                                            w1f0, w2f0, fused, g0);

    // GIN layer 1
    sfg_zero_f32<<<4096, 256, 0, stream>>>(agg, (int)HB);
    sfg_scatter <<<(NEDGES * 16) / 256, 256, 0, stream>>>(g0, ei, agg);
    sfg_gin_dense<<<TILES, 32, 0, stream>>>(g0, agg, g0, newx,
                                            w1f1, w2f1, fused + 256, g1);

    // pooling + head
    sfg_zero_f32<<<33, 256, 0, stream>>>(sums, NGRAPH * HDIM + NGRAPH);
    sfg_pool<<<(NNODES * 16 + 255) / 256, 256, 0, stream>>>(g1, batch, sums, cnts);
    sfg_head<<<4, 32, 0, stream>>>(sums, cnts, q1w, q1b, q2w, q2b, out);
}
